// Encoder_22316650070699
// MI455X (gfx1250) — matrix operand order repaired and checked by reading, not compiled, not run on hardware
//
#include <hip/hip_runtime.h>
#include <hip/hip_bf16.h>

// ---------------- types ----------------
typedef __bf16 bf16;
typedef __attribute__((ext_vector_type(16))) __bf16 v16bf;
typedef __attribute__((ext_vector_type(8)))  float  v8f;
typedef __attribute__((ext_vector_type(4)))  unsigned int v4u;

union ABFrag { v16bf v; v4u q[2]; };
union CFrag  { v8f v;  float f[8]; };

#define Bc 8
#define Sc 1024
#define Dc 768
#define Hc 12
#define HDc 64
#define DFc 3072
#define Lc 6
#define Mrows (Bc*Sc)   // 8192

__device__ __forceinline__ v8f wmma_bf16(v16bf a, v16bf b, v8f c) {
    return __builtin_amdgcn_wmma_f32_16x16x32_bf16(false, a, false, b, (short)0, c, false, false);
}

// ---- crosslane butterfly within 16-lane rows (VALU v_permlane16_b32) ----
#if __has_builtin(__builtin_amdgcn_permlane16)
__device__ __forceinline__ float perm16f(float v, unsigned lo, unsigned hi) {
    unsigned i = __builtin_bit_cast(unsigned, v);
    unsigned r = __builtin_amdgcn_permlane16(i, i, lo, hi, false, false);
    return __builtin_bit_cast(float, r);
}
__device__ __forceinline__ float rowmax16(float v) {
    v = fmaxf(v, perm16f(v, 0xFEDCBA98u, 0x76543210u)); // xor 8
    v = fmaxf(v, perm16f(v, 0x32107654u, 0xBA98FEDCu)); // xor 4
    v = fmaxf(v, perm16f(v, 0x54761032u, 0xDCFE98BAu)); // xor 2
    v = fmaxf(v, perm16f(v, 0x67452301u, 0xEFCDAB89u)); // xor 1
    return v;
}
__device__ __forceinline__ float rowsum16(float v) {
    v += perm16f(v, 0xFEDCBA98u, 0x76543210u);
    v += perm16f(v, 0x32107654u, 0xBA98FEDCu);
    v += perm16f(v, 0x54761032u, 0xDCFE98BAu);
    v += perm16f(v, 0x67452301u, 0xEFCDAB89u);
    return v;
}
#else
__device__ __forceinline__ float rowmax16(float v) {
#pragma unroll
    for (int m = 8; m >= 1; m >>= 1) v = fmaxf(v, __shfl_xor(v, m, 32));
    return v;
}
__device__ __forceinline__ float rowsum16(float v) {
#pragma unroll
    for (int m = 8; m >= 1; m >>= 1) v += __shfl_xor(v, m, 32);
    return v;
}
#endif

// ---------------- fp32 -> bf16 convert ----------------
__global__ __launch_bounds__(256) void k_f2b(const float* __restrict__ in,
                                             bf16* __restrict__ out, int n) {
    int i = blockIdx.x * 256 + threadIdx.x;
    if (i < n) out[i] = (bf16)in[i];
}

// ---------------- embedding + positional encoding ----------------
__global__ __launch_bounds__(256) void k_embed(const int* __restrict__ ids,
                                               const float* __restrict__ emb,
                                               const float* __restrict__ pe,
                                               float* __restrict__ x,
                                               bf16* __restrict__ xb) {
    int row = blockIdx.x;            // b*S + s
    int tid = threadIdx.x;
    int id  = ids[row];
    int s   = row & (Sc - 1);
#pragma unroll
    for (int i = 0; i < 3; i++) {
        int e = tid + i * 256;
        float v = emb[(size_t)id * Dc + e] + pe[(size_t)s * Dc + e];
        x [(size_t)row * Dc + e] = v;
        xb[(size_t)row * Dc + e] = (bf16)v;
    }
}

// ---------------- GEMM: C[M,N] = A[M,K] @ W[N,K]^T + bias ----------------
// Each wave computes a 32x64 tile: 2 A frags + 4 shared B frags -> 8 WMMAs/k-step.
// FLAGS: 1 = relu, 2 = store fp32 Cf, 4 = store bf16 Cb
template <int FLAGS>
__global__ __launch_bounds__(256) void k_gemm(const bf16* __restrict__ A,
                                              const bf16* __restrict__ W,
                                              const float* __restrict__ bias,
                                              float* __restrict__ Cf,
                                              bf16* __restrict__ Cb,
                                              int Mm, int Nn, int Kk) {
    const int wave = threadIdx.x >> 5;
    const int lid  = threadIdx.x & 31;
    const int hh   = lid >> 4;     // half of wave
    const int ln   = lid & 15;     // row (A) / col (B,C)
    int t  = blockIdx.x * 8 + wave;
    int ng = t % (Nn >> 6);        // 64-wide column group
    int mt = t / (Nn >> 6);        // 32-row tile

    const bf16* Arow0 = A + (size_t)(mt * 32 + ln) * Kk;
    const bf16* Arow1 = Arow0 + (size_t)16 * Kk;
    const bf16* Wr[4];
#pragma unroll
    for (int u = 0; u < 4; u++)
        Wr[u] = W + (size_t)(ng * 64 + u * 16 + ln) * Kk + 8 * hh;

    CFrag acc[2][4] = {};
    for (int k0 = 0; k0 < Kk; k0 += 32) {
        ABFrag a0, a1;
        a0.q[0] = *(const v4u*)(Arow0 + k0 + 8 * hh);
        a0.q[1] = *(const v4u*)(Arow0 + k0 + 16 + 8 * hh);
        a1.q[0] = *(const v4u*)(Arow1 + k0 + 8 * hh);
        a1.q[1] = *(const v4u*)(Arow1 + k0 + 16 + 8 * hh);
#pragma unroll
        for (int u = 0; u < 4; u++) {
            ABFrag b;
            b.q[0] = *(const v4u*)(Wr[u] + k0);
            b.q[1] = *(const v4u*)(Wr[u] + k0 + 16);
            acc[0][u].v = wmma_bf16(a0.v, b.v, acc[0][u].v);
            acc[1][u].v = wmma_bf16(a1.v, b.v, acc[1][u].v);
        }
    }
#pragma unroll
    for (int r2 = 0; r2 < 2; r2++)
#pragma unroll
        for (int u = 0; u < 4; u++) {
            float bv = bias[ng * 64 + u * 16 + ln];
#pragma unroll
            for (int j = 0; j < 8; j++) {
                float v = acc[r2][u].f[j] + bv;
                if (FLAGS & 1) v = fmaxf(v, 0.0f);
                size_t idx = (size_t)(mt * 32 + r2 * 16 + j + 8 * hh) * Nn
                           + ng * 64 + u * 16 + ln;
                if (FLAGS & 2) Cf[idx] = v;
                if (FLAGS & 4) Cb[idx] = (bf16)v;
            }
        }
}

// ---------------- V transpose via LDS tile: vt[b,h,d,s] = v[b,s,h*64+d] ----------------
__global__ __launch_bounds__(256) void k_transpose_v(const bf16* __restrict__ vb,
                                                     bf16* __restrict__ vt) {
    __shared__ bf16 tile[64][72];            // padded rows
    int blk  = blockIdx.x;                   // bh*(S/64) + sblk
    int sblk = blk & 15;
    int bh   = blk >> 4;
    int b = bh / Hc, h = bh % Hc;
    int t = threadIdx.x;
    int s0 = sblk * 64;
#pragma unroll
    for (int i = 0; i < 16; i++) {           // coalesced along d
        int e = t + 256 * i;
        int d = e & 63, s = e >> 6;
        tile[d][s] = vb[(size_t)(b * Sc + s0 + s) * Dc + h * HDc + d];
    }
    __syncthreads();
#pragma unroll
    for (int i = 0; i < 16; i++) {           // coalesced along s
        int e = t + 256 * i;
        int s = e & 63, d = e >> 6;
        vt[(size_t)(bh * HDc + d) * Sc + s0 + s] = tile[d][s];
    }
}

// ---------------- flash attention: one 16-row q-tile per wave ----------------
__global__ __launch_bounds__(256) void k_flash(const bf16* __restrict__ qb,
                                               const bf16* __restrict__ kb,
                                               const bf16* __restrict__ vt,
                                               float* __restrict__ out) {
    __shared__ __align__(16) bf16 pbuf[8][16 * 32];
    const int wave = threadIdx.x >> 5;
    const int lid  = threadIdx.x & 31;
    const int hh   = lid >> 4;
    const int ln   = lid & 15;

    int qt   = blockIdx.x * 8 + wave;       // global q-tile id
    int bh   = qt >> 6;                     // 64 q-tiles per (b,h)
    int q0   = (qt & 63) << 4;
    int b    = bh / Hc, head = bh % Hc;

    // Q fragments (hd 0..31 and 32..63), resident for whole pass
    const bf16* Qrow = qb + (size_t)(b * Sc + q0 + ln) * Dc + head * HDc;
    ABFrag qa[2];
#pragma unroll
    for (int kk = 0; kk < 2; kk++) {
        qa[kk].q[0] = *(const v4u*)(Qrow + 32 * kk + 8 * hh);
        qa[kk].q[1] = *(const v4u*)(Qrow + 32 * kk + 16 + 8 * hh);
    }

    float mrow[8], lrow[8];
    CFrag o[4];
#pragma unroll
    for (int j = 0; j < 8; j++) { mrow[j] = -1e30f; lrow[j] = 0.0f; }
#pragma unroll
    for (int nt = 0; nt < 4; nt++)
#pragma unroll
        for (int j = 0; j < 8; j++) o[nt].f[j] = 0.0f;

    bf16* pw = &pbuf[wave][0];

    for (int kt0 = 0; kt0 < Sc; kt0 += 32) {
        // ---- scores S = (Q K^T) * 1/sqrt(64) for 32 keys (two 16-col frags)
        CFrag sf[2];
#pragma unroll
        for (int f = 0; f < 2; f++) {
            const bf16* Krow = kb + (size_t)(b * Sc + kt0 + 16 * f + ln) * Dc + head * HDc;
            v8f acc = {};
#pragma unroll
            for (int kk = 0; kk < 2; kk++) {
                ABFrag kf;
                kf.q[0] = *(const v4u*)(Krow + 32 * kk + 8 * hh);
                kf.q[1] = *(const v4u*)(Krow + 32 * kk + 16 + 8 * hh);
                acc = wmma_bf16(qa[kk].v, kf.v, acc);
            }
            sf[f].v = acc;
#pragma unroll
            for (int j = 0; j < 8; j++) sf[f].f[j] *= 0.125f;
        }
        // ---- online softmax statistics (rows live across 16-lane halves)
        float mnew[8], alpha[8];
#pragma unroll
        for (int j = 0; j < 8; j++) {
            float mx = rowmax16(fmaxf(sf[0].f[j], sf[1].f[j]));
            mnew[j]  = fmaxf(mrow[j], mx);
            alpha[j] = __expf(mrow[j] - mnew[j]);
            mrow[j]  = mnew[j];
        }
        // ---- P = exp(S - m), stage to LDS for C->A relayout
#pragma unroll
        for (int f = 0; f < 2; f++)
#pragma unroll
            for (int j = 0; j < 8; j++) {
                float p = __expf(sf[f].f[j] - mnew[j]);
                sf[f].f[j] = p;
                pw[(j + 8 * hh) * 32 + 16 * f + ln] = (bf16)p;
            }
#pragma unroll
        for (int j = 0; j < 8; j++) {
            float rs = rowsum16(sf[0].f[j] + sf[1].f[j]);
            lrow[j] = lrow[j] * alpha[j] + rs;
        }
        asm volatile("s_wait_dscnt 0x0" ::: "memory");
        ABFrag pa;   // P as 16x32 A-operand
        pa.q[0] = *(const v4u*)(pw + ln * 32 + 8 * hh);
        pa.q[1] = *(const v4u*)(pw + ln * 32 + 16 + 8 * hh);
        // ---- O = O*alpha + P @ V   (Vt rows are hd, contiguous along s)
#pragma unroll
        for (int nt = 0; nt < 4; nt++) {
#pragma unroll
            for (int j = 0; j < 8; j++) o[nt].f[j] *= alpha[j];
            const bf16* Vrow = vt + (size_t)(bh * HDc + nt * 16 + ln) * Sc + kt0;
            ABFrag vf;
            vf.q[0] = *(const v4u*)(Vrow + 8 * hh);
            vf.q[1] = *(const v4u*)(Vrow + 16 + 8 * hh);
            o[nt].v = wmma_bf16(pa.v, vf.v, o[nt].v);
        }
    }
    // ---- normalize and store fp32 attention output in [B*S, D] layout
    float rl[8];
#pragma unroll
    for (int j = 0; j < 8; j++) rl[j] = 1.0f / lrow[j];
#pragma unroll
    for (int nt = 0; nt < 4; nt++)
#pragma unroll
        for (int j = 0; j < 8; j++)
            out[(size_t)(b * Sc + q0 + j + 8 * hh) * Dc + head * HDc + nt * 16 + ln] =
                o[nt].f[j] * rl[j];
}

// ---------------- residual add + layernorm ----------------
__device__ __forceinline__ float blockSum(float v, volatile float* sbuf) {
#pragma unroll
    for (int m = 16; m >= 1; m >>= 1) v += __shfl_xor(v, m, 32);
    int w = threadIdx.x >> 5, l = threadIdx.x & 31;
    __syncthreads();
    if (l == 0) sbuf[w] = v;
    __syncthreads();
    float t = sbuf[0];
#pragma unroll
    for (int i = 1; i < 8; i++) t += sbuf[i];
    return t;
}

__global__ __launch_bounds__(256) void k_add_ln(const float* __restrict__ x,
                                                const float* __restrict__ r,
                                                const float* __restrict__ g,
                                                const float* __restrict__ bta,
                                                float* __restrict__ y,
                                                bf16* __restrict__ yb) {
    __shared__ float sbuf[8];
    int row = blockIdx.x, tid = threadIdx.x;
    float v[3], s = 0.0f, sq = 0.0f;
#pragma unroll
    for (int i = 0; i < 3; i++) {
        int e = tid + i * 256;
        v[i] = x[(size_t)row * Dc + e] + r[(size_t)row * Dc + e];
        s += v[i]; sq += v[i] * v[i];
    }
    s  = blockSum(s,  sbuf);
    sq = blockSum(sq, sbuf);
    float mu   = s * (1.0f / Dc);
    float var  = sq * (1.0f / Dc) - mu * mu;
    float rstd = rsqrtf(var + 1e-5f);
#pragma unroll
    for (int i = 0; i < 3; i++) {
        int e = tid + i * 256;
        float o = (v[i] - mu) * rstd * g[e] + bta[e];
        y [(size_t)row * Dc + e] = o;
        yb[(size_t)row * Dc + e] = (bf16)o;
    }
}

// ---------------- host orchestration ----------------
extern "C" void kernel_launch(void* const* d_in, const int* in_sizes, int n_in,
                              void* d_out, int out_size, void* d_ws, size_t ws_size,
                              hipStream_t stream) {
    const int*   ids   = (const int*)  d_in[0];
    const float* emb   = (const float*)d_in[1];
    const float* pe    = (const float*)d_in[2];
    const float* Wq    = (const float*)d_in[3];
    const float* bq    = (const float*)d_in[4];
    const float* Wk    = (const float*)d_in[5];
    const float* bk    = (const float*)d_in[6];
    const float* Wv    = (const float*)d_in[7];
    const float* bv    = (const float*)d_in[8];
    const float* W1    = (const float*)d_in[9];
    const float* b1    = (const float*)d_in[10];
    const float* W2    = (const float*)d_in[11];
    const float* b2    = (const float*)d_in[12];
    const float* ln1g  = (const float*)d_in[13];
    const float* ln1b  = (const float*)d_in[14];
    const float* ln2g  = (const float*)d_in[15];
    const float* ln2b  = (const float*)d_in[16];
    float* xout = (float*)d_out;                 // x lives in d_out across layers

    char* ws = (char*)d_ws;
    size_t off = 0;
    auto take = [&](size_t bytes) { size_t r = off; off += (bytes + 255) & ~(size_t)255; return ws + r; };

    const size_t MD = (size_t)Mrows * Dc;
    bf16*  xb  = (bf16*) take(MD * 2);
    bf16*  qb  = (bf16*) take(MD * 2);
    bf16*  kbf = (bf16*) take(MD * 2);
    bf16*  vbf = (bf16*) take(MD * 2);
    bf16*  vt  = (bf16*) take(MD * 2);           // B*H*64*S == M*D elements
    float* y1  = (float*)take(MD * 4);
    bf16*  y1b = (bf16*) take(MD * 2);
    float* hb  = (float*)take(MD * 4);           // h1, reused for h2
    bf16*  fb  = (bf16*) take((size_t)Mrows * DFc * 2);
    bf16*  wqb = (bf16*) take((size_t)Dc * Dc * 2);
    bf16*  wkb = (bf16*) take((size_t)Dc * Dc * 2);
    bf16*  wvb = (bf16*) take((size_t)Dc * Dc * 2);
    bf16*  w1b = (bf16*) take((size_t)DFc * Dc * 2);
    bf16*  w2b = (bf16*) take((size_t)Dc * DFc * 2);

    k_embed<<<Mrows, 256, 0, stream>>>(ids, emb, pe, xout, xb);

    const int gQKV = (Mrows / 32) * (Dc  / 64) / 8;   // 384
    const int gFF1 = (Mrows / 32) * (DFc / 64) / 8;   // 1536
    const int nDD  = Dc * Dc, nFD = DFc * Dc;

    for (int l = 0; l < Lc; l++) {
        const float* Wq_l = Wq + (size_t)l * nDD;
        const float* Wk_l = Wk + (size_t)l * nDD;
        const float* Wv_l = Wv + (size_t)l * nDD;
        const float* W1_l = W1 + (size_t)l * nFD;
        const float* W2_l = W2 + (size_t)l * nFD;

        k_f2b<<<nDD / 256, 256, 0, stream>>>(Wq_l, wqb, nDD);
        k_f2b<<<nDD / 256, 256, 0, stream>>>(Wk_l, wkb, nDD);
        k_f2b<<<nDD / 256, 256, 0, stream>>>(Wv_l, wvb, nDD);
        k_f2b<<<nFD / 256, 256, 0, stream>>>(W1_l, w1b, nFD);
        k_f2b<<<nFD / 256, 256, 0, stream>>>(W2_l, w2b, nFD);

        k_gemm<4><<<gQKV, 256, 0, stream>>>(xb, wqb, bq + l * Dc, nullptr, qb,  Mrows, Dc, Dc);
        k_gemm<4><<<gQKV, 256, 0, stream>>>(xb, wkb, bk + l * Dc, nullptr, kbf, Mrows, Dc, Dc);
        k_gemm<4><<<gQKV, 256, 0, stream>>>(xb, wvb, bv + l * Dc, nullptr, vbf, Mrows, Dc, Dc);

        k_transpose_v<<<Bc * Hc * (Sc / 64), 256, 0, stream>>>(vbf, vt);
        k_flash<<<Bc * Hc * (Sc / 16) / 8, 256, 0, stream>>>(qb, kbf, vt, hb);

        k_add_ln<<<Mrows, 256, 0, stream>>>(xout, hb, ln1g + l * Dc, ln1b + l * Dc, y1, y1b);

        k_gemm<5><<<gFF1, 256, 0, stream>>>(y1b, w1b, b1 + l * DFc, nullptr, fb, Mrows, DFc, Dc);
        k_gemm<2><<<gQKV, 256, 0, stream>>>(fb,  w2b, b2 + l * Dc,  hb, nullptr, Mrows, Dc, DFc);

        k_add_ln<<<Mrows, 256, 0, stream>>>(y1, hb, ln2g + l * Dc, ln2b + l * Dc, xout, xb);
    }
}
